// ImagingEncoder_70798240907608
// MI455X (gfx1250) — matrix operand order repaired and checked by reading, not compiled, not run on hardware
//
#include <hip/hip_runtime.h>
#include <hip/hip_bf16.h>

typedef __attribute__((ext_vector_type(16))) __bf16 v16bf;
typedef __attribute__((ext_vector_type(8)))  __bf16 v8bf;
typedef __attribute__((ext_vector_type(8)))  float  v8f;

static constexpr int NB = 16;    // batch
static constexpr int H_ = 256;
static constexpr int W_ = 256;

// ---------------------------------------------------------------------------
// Build the 30-filter SRM 5x5 bank (Fridrich) into workspace: srm[30][5][5].
// ---------------------------------------------------------------------------
__global__ void build_srm(float* __restrict__ srm) {
  if (threadIdx.x != 0 || blockIdx.x != 0) return;
  for (int i = 0; i < 750; ++i) srm[i] = 0.f;
  const int D8[8][2] = {{-1,0},{-1,1},{0,1},{1,1},{1,0},{1,-1},{0,-1},{-1,-1}};
  int f = 0;
  for (int i = 0; i < 8; ++i) {               // 1st order
    int dy = D8[i][0], dx = D8[i][1];
    srm[f*25 + 12] = -1.f;
    srm[f*25 + (2+dy)*5 + (2+dx)] = 1.f;
    ++f;
  }
  for (int i = 0; i < 4; ++i) {               // 2nd order /2
    int dy = D8[i][0], dx = D8[i][1];
    srm[f*25 + 12] = -1.f;
    srm[f*25 + (2+dy)*5 + (2+dx)] = 0.5f;
    srm[f*25 + (2-dy)*5 + (2-dx)] = 0.5f;
    ++f;
  }
  for (int i = 0; i < 8; ++i) {               // 3rd order /3
    int dy = D8[i][0], dx = D8[i][1];
    srm[f*25 + (2-dy)*5 + (2-dx)]     = -1.f/3.f;
    srm[f*25 + 12]                    =  1.f;
    srm[f*25 + (2+dy)*5 + (2+dx)]     = -1.f;
    srm[f*25 + (2+2*dy)*5 + (2+2*dx)] =  1.f/3.f;
    ++f;
  }
  {                                           // SQUARE 3x3 /4
    const float sq3[9] = {-1,2,-1, 2,-4,2, -1,2,-1};
    for (int i = 0; i < 3; ++i)
      for (int j = 0; j < 3; ++j)
        srm[f*25 + (1+i)*5 + (1+j)] = sq3[i*3+j] * 0.25f;
    ++f;
  }
  {                                           // SQUARE 5x5 /12
    const float sq5[25] = {-1,2,-2,2,-1, 2,-6,8,-6,2, -2,8,-12,8,-2,
                           2,-6,8,-6,2, -1,2,-2,2,-1};
    for (int i = 0; i < 25; ++i) srm[f*25 + i] = sq5[i] / 12.f;
    ++f;
  }
  {                                           // EDGE 3x3 rot /4 ; rot90 CCW
    const float e3[9] = {-1,2,-1, 2,-4,2, 0,0,0};
    for (int r = 0; r < 4; ++r) {
      float cur[9]; for (int i = 0; i < 9; ++i) cur[i] = e3[i] * 0.25f;
      for (int t = 0; t < r; ++t) {
        float tmp[9];
        for (int i = 0; i < 3; ++i)
          for (int j = 0; j < 3; ++j)
            tmp[i*3+j] = cur[j*3 + (2-i)];
        for (int i = 0; i < 9; ++i) cur[i] = tmp[i];
      }
      for (int i = 0; i < 3; ++i)
        for (int j = 0; j < 3; ++j)
          srm[f*25 + (1+i)*5 + (1+j)] = cur[i*3+j];
      ++f;
    }
  }
  {                                           // EDGE 5x5 rot /4
    const float e5[25] = {-1,2,-2,2,-1, 2,-6,8,-6,2, -2,8,-12,8,-2,
                          0,0,0,0,0, 0,0,0,0,0};
    for (int r = 0; r < 4; ++r) {
      float cur[25]; for (int i = 0; i < 25; ++i) cur[i] = e5[i] * 0.25f;
      for (int t = 0; t < r; ++t) {
        float tmp[25];
        for (int i = 0; i < 5; ++i)
          for (int j = 0; j < 5; ++j)
            tmp[i*5+j] = cur[j*5 + (4-i)];
        for (int i = 0; i < 25; ++i) cur[i] = tmp[i];
      }
      for (int i = 0; i < 25; ++i) srm[f*25 + i] = cur[i];
      ++f;
    }
  }
}

// ---------------------------------------------------------------------------
// Fold BN into conv weights; pre-pack WMMA B-fragment order:
//   wpk[kstep][ngroup][lane][16 bf16]   K = ky*3*CIN + kx*CIN + ci
// B-fragment lane map: half=lane>>4, n=lane&15; element e -> K_local=half*16+e.
// ---------------------------------------------------------------------------
template<int CIN>
__global__ void pack_weights(const float* __restrict__ w,     // [64][CIN][3][3]
                             const float* __restrict__ bconv,
                             const float* __restrict__ gam,
                             const float* __restrict__ bet,
                             const float* __restrict__ mu,
                             const float* __restrict__ var,
                             __bf16* __restrict__ wpk,
                             float*  __restrict__ bias) {
  constexpr int KS = (9 * CIN) / 32;
  const int tid = blockIdx.x * 256 + threadIdx.x;
  if (tid < 64) {
    const float sc = gam[tid] * rsqrtf(var[tid] + 1e-5f);
    bias[tid] = (bconv[tid] - mu[tid]) * sc + bet[tid];
  }
  if (tid >= KS * 2048) return;
  const int e    = tid & 15;
  const int lane = (tid >> 4) & 31;
  const int gq   = (tid >> 9) & 3;
  const int s    = tid >> 11;
  const int half = lane >> 4;
  const int nl   = lane & 15;
  const int N    = gq * 16 + nl;
  const int k    = s * 32 + half * 16 + e;
  const int ky   = k / (3 * CIN);
  const int kx   = (k / CIN) % 3;
  const int ci   = k % CIN;
  const float sc = gam[N] * rsqrtf(var[N] + 1e-5f);
  const float wv = w[(((size_t)N * CIN + ci) * 3 + ky) * 3 + kx] * sc;
  wpk[tid] = (__bf16)wv;
}

// ---------------------------------------------------------------------------
// Feature extraction -> NHWC bf16 [B,H,W,32]: ch0=prnu, 1..30=srm, 31=cfa.
// 16x16 tile; gray/green with 2-px halo in LDS; 4x uint4 vector store/pixel.
// ---------------------------------------------------------------------------
__global__ __launch_bounds__(256) void feature_kernel(
    const float* __restrict__ x,    // NCHW f32 [16,3,256,256]
    const float* __restrict__ srm,  // [30][5][5]
    __bf16* __restrict__ feat) {
  __shared__ float sgray[20 * 20];
  __shared__ float sgreen[20 * 20];
  __shared__ float s_srm[750];

  const int t  = blockIdx.x & 255;
  const int b  = blockIdx.x >> 8;
  const int y0 = (t >> 4) * 16;
  const int x0 = (t & 15) * 16;

  for (int i = threadIdx.x; i < 750; i += 256) s_srm[i] = srm[i];

  for (int i = threadIdx.x; i < 400; i += 256) {
    const int r = i / 20, c = i % 20;
    const int gy = y0 - 2 + r, gx = x0 - 2 + c;
    float gray = 0.f, green = 0.f;
    if ((unsigned)gy < (unsigned)H_ && (unsigned)gx < (unsigned)W_) {
      float rr = x[(((size_t)b * 3 + 0) * H_ + gy) * W_ + gx] * 0.229f + 0.485f;
      float gg = x[(((size_t)b * 3 + 1) * H_ + gy) * W_ + gx] * 0.224f + 0.456f;
      float bb = x[(((size_t)b * 3 + 2) * H_ + gy) * W_ + gx] * 0.225f + 0.406f;
      rr = fminf(fmaxf(rr, 0.f), 1.f);
      gg = fminf(fmaxf(gg, 0.f), 1.f);
      bb = fminf(fmaxf(bb, 0.f), 1.f);
      gray  = 0.299f * rr + 0.587f * gg + 0.114f * bb;
      green = gg;
    }
    sgray[i]  = gray;
    sgreen[i] = green;
  }
  __syncthreads();

  const int py = threadIdx.x >> 4, px = threadIdx.x & 15;

  float wbuf[25];
#pragma unroll
  for (int i = 0; i < 5; ++i)
#pragma unroll
    for (int j = 0; j < 5; ++j)
      wbuf[i * 5 + j] = sgray[(py + i) * 20 + (px + j)];

  const float g1d[5] = {1.f, 4.f, 6.f, 4.f, 1.f};
  float gs = 0.f;
#pragma unroll
  for (int i = 0; i < 5; ++i)
#pragma unroll
    for (int j = 0; j < 5; ++j)
      gs += g1d[i] * g1d[j] * wbuf[i * 5 + j];

  const float gcen = sgreen[(py + 2) * 20 + (px + 2)];
  const float cfa  = gcen - 0.25f * (sgreen[(py + 1) * 20 + (px + 2)] +
                                     sgreen[(py + 3) * 20 + (px + 2)] +
                                     sgreen[(py + 2) * 20 + (px + 1)] +
                                     sgreen[(py + 2) * 20 + (px + 3)]);

  __bf16 ov[32];
  ov[0]  = (__bf16)(wbuf[12] - gs * (1.f / 256.f));   // prnu
#pragma unroll
  for (int f = 0; f < 30; ++f) {
    float s = 0.f;
#pragma unroll
    for (int q = 0; q < 25; ++q) s += s_srm[f * 25 + q] * wbuf[q];
    ov[1 + f] = (__bf16)s;
  }
  ov[31] = (__bf16)cfa;

  __bf16* dst = &feat[(((size_t)b * H_ + (y0 + py)) * W_ + (x0 + px)) * 32];
  const uint4* s4 = (const uint4*)ov;
  uint4* d4 = (uint4*)dst;
#pragma unroll
  for (int q = 0; q < 4; ++q) d4[q] = s4[q];
}

// ---------------------------------------------------------------------------
// Fused 3x3 conv + BN(folded) + ReLU, implicit GEMM via WMMA bf16 / f32 acc.
// Block tile: 4 rows x 32 cols x 64 out-ch (8 waves; 2 waves per row, M=16).
// Halo tile 6x34xCIN (padded pixel stride CIN+8 -> conflict-free A loads) and
// all packed weights staged in LDS (shared by 8 waves, removes VMEM->WMMA
// dependency chain). Epilogue for bf16 layers routes C tiles through LDS for
// fully coalesced b128 stores; final layer writes f32 NCHW directly
// (8 consecutive-W floats per lane).
// ---------------------------------------------------------------------------
template<int CIN, bool FINAL>
__global__ __launch_bounds__(256) void conv3x3_wmma(
    const __bf16* __restrict__ in,   // NHWC bf16 [16,256,256,CIN]
    const __bf16* __restrict__ wpk,  // [KS][4][32][16] bf16 (B fragments)
    const float*  __restrict__ bias, // [64]
    void* __restrict__ outv) {
  constexpr int KS   = (9 * CIN) / 32;
  constexpr int PSTR = CIN + 8;              // padded pixel stride (elements)
  constexpr int TILE_ELEMS = 6 * 34 * PSTR;
  constexpr int SOUT_ELEMS = 128 * 72;       // 4x32 px, 64 ch + 8 pad
  constexpr int SMEM_ELEMS = (TILE_ELEMS > SOUT_ELEMS) ? TILE_ELEMS : SOUT_ELEMS;
  __shared__ __bf16 smem[SMEM_ELEMS];
  __shared__ __bf16 swt[KS * 2048];

  const int wt = blockIdx.x & 7;             // W/32 = 8
  const int h0 = ((blockIdx.x >> 3) & 63) * 4;
  const int b  = blockIdx.x >> 9;
  const int w0 = wt * 32;

  // ---- stage packed weights into LDS (coalesced b128) ----
  {
    const uint4* src = (const uint4*)wpk;
    uint4* dst = (uint4*)swt;
    for (int i = threadIdx.x; i < (KS * 2048) / 8; i += 256) dst[i] = src[i];
  }
  // ---- stage input halo tile 6 x 34 x CIN (zero pad OOB) ----
  for (int i = threadIdx.x; i < 6 * 34; i += 256) {
    const int r  = i / 34, c = i % 34;
    const int gr = h0 - 1 + r, gc = w0 - 1 + c;
    uint4* dst = (uint4*)&smem[(size_t)(r * 34 + c) * PSTR];
    if ((unsigned)gr < (unsigned)H_ && (unsigned)gc < (unsigned)W_) {
      const uint4* src = (const uint4*)&in[(((size_t)b * H_ + gr) * W_ + gc) * CIN];
#pragma unroll
      for (int q = 0; q < CIN / 8; ++q) dst[q] = src[q];
    } else {
      const uint4 z = make_uint4(0, 0, 0, 0);
#pragma unroll
      for (int q = 0; q < CIN / 8; ++q) dst[q] = z;
    }
  }
  __syncthreads();

  const int lane = threadIdx.x & 31;
  const int wave = threadIdx.x >> 5;
  const int rw   = wave >> 1;                // output row within tile, 0..3
  const int m0   = (wave & 1) * 16;          // col offset within 32-wide seg
  const int half = lane >> 4;
  const int nl   = lane & 15;

  v8f acc[4] = {};

#pragma unroll
  for (int s = 0; s < KS; ++s) {
    const int k0  = s * 32;
    const int ky  = k0 / (3 * CIN);
    const int kx  = (k0 / CIN) % 3;
    const int ci0 = k0 % CIN;

    // A fragment: lane row M=nl; K_local = half*16 + e
    const __bf16* ap =
        &smem[(size_t)((rw + ky) * 34 + (m0 + nl + kx)) * PSTR + ci0 + half * 16];
    const v8bf alo = *(const v8bf*)ap;
    const v8bf ahi = *(const v8bf*)(ap + 8);
    const v16bf afrag = __builtin_shufflevector(
        alo, ahi, 0, 1, 2, 3, 4, 5, 6, 7, 8, 9, 10, 11, 12, 13, 14, 15);

    const v16bf* wp = (const v16bf*)swt + (s * 4) * 32 + lane;
#pragma unroll
    for (int g = 0; g < 4; ++g) {
      const v16bf bfrag = wp[g * 32];
      acc[g] = __builtin_amdgcn_wmma_f32_16x16x32_bf16(
          false, afrag, false, bfrag, (short)0, acc[g], false, false);
    }
  }

  const int h = h0 + rw;
  if (FINAL) {
    // f32 NCHW direct: 8 consecutive W positions per lane per group
    float* out = (float*)outv;
#pragma unroll
    for (int g = 0; g < 4; ++g) {
      const float bb = bias[g * 16 + nl];
      const int ch   = g * 16 + nl;
      float* p = &out[(((size_t)b * 64 + ch) * H_ + h) * W_ + w0 + m0 + half * 8];
#pragma unroll
      for (int r = 0; r < 8; ++r) {
        const float v = acc[g][r] + bb;
        p[r] = v > 0.f ? v : 0.f;
      }
    }
  } else {
    // bf16 NHWC: transpose through LDS, then coalesced b128 stores
    __syncthreads();   // all waves done reading the input tile
#pragma unroll
    for (int g = 0; g < 4; ++g) {
      const float bb = bias[g * 16 + nl];
#pragma unroll
      for (int r = 0; r < 8; ++r) {
        const int px = rw * 32 + m0 + half * 8 + r;
        float v = acc[g][r] + bb;
        v = v > 0.f ? v : 0.f;
        smem[(size_t)px * 72 + g * 16 + nl] = (__bf16)v;
      }
    }
    __syncthreads();
    __bf16* out = (__bf16*)outv;
    for (int u = threadIdx.x; u < 1024; u += 256) {   // 128 px * 8 uint4
      const int px = u >> 3, q = u & 7;
      const int pr = px >> 5, pc = px & 31;
      ((uint4*)&out[(((size_t)b * H_ + h0 + pr) * W_ + w0 + pc) * 64])[q] =
          ((const uint4*)&smem[(size_t)px * 72])[q];
    }
  }
}

// ---------------------------------------------------------------------------
extern "C" void kernel_launch(void* const* d_in, const int* in_sizes, int n_in,
                              void* d_out, int out_size, void* d_ws,
                              size_t ws_size, hipStream_t stream) {
  const float* x = (const float*)d_in[0];
  const float *Wc[4], *Bc[4], *Gc[4], *Be[4], *Mu[4], *Va[4];
  for (int i = 0; i < 4; ++i) {
    Wc[i] = (const float*)d_in[1 + 6 * i + 0];
    Bc[i] = (const float*)d_in[1 + 6 * i + 1];
    Gc[i] = (const float*)d_in[1 + 6 * i + 2];
    Be[i] = (const float*)d_in[1 + 6 * i + 3];
    Mu[i] = (const float*)d_in[1 + 6 * i + 4];
    Va[i] = (const float*)d_in[1 + 6 * i + 5];
  }

  char* ws = (char*)d_ws;
  size_t off = 0;
  float* srm = (float*)(ws + off); off += 4096;
  __bf16* feat = (__bf16*)(ws + off); off += (size_t)NB * H_ * W_ * 32 * 2;  // 64 MB
  __bf16* buf0 = (__bf16*)(ws + off); off += (size_t)NB * H_ * W_ * 64 * 2;  // 128 MB
  __bf16* buf1 = (__bf16*)(ws + off); off += (size_t)NB * H_ * W_ * 64 * 2;  // 128 MB
  __bf16* wp[4];
  float*  bi[4];
  const size_t wpsz[4] = {(size_t)9 * 2048 * 2, (size_t)18 * 2048 * 2,
                          (size_t)18 * 2048 * 2, (size_t)18 * 2048 * 2};
  for (int i = 0; i < 4; ++i) { wp[i] = (__bf16*)(ws + off); off += wpsz[i]; }
  for (int i = 0; i < 4; ++i) { bi[i] = (float*)(ws + off);  off += 256; }

  build_srm<<<1, 64, 0, stream>>>(srm);

  pack_weights<32><<<(9 * 2048 + 255) / 256, 256, 0, stream>>>(
      Wc[0], Bc[0], Gc[0], Be[0], Mu[0], Va[0], wp[0], bi[0]);
  for (int i = 1; i < 4; ++i)
    pack_weights<64><<<(18 * 2048 + 255) / 256, 256, 0, stream>>>(
        Wc[i], Bc[i], Gc[i], Be[i], Mu[i], Va[i], wp[i], bi[i]);

  feature_kernel<<<NB * 256, 256, 0, stream>>>(x, srm, feat);

  const int nblk = NB * (H_ / 4) * (W_ / 32);  // 8192
  conv3x3_wmma<32, false><<<nblk, 256, 0, stream>>>(feat, wp[0], bi[0], buf0);
  conv3x3_wmma<64, false><<<nblk, 256, 0, stream>>>(buf0, wp[1], bi[1], buf1);
  conv3x3_wmma<64, false><<<nblk, 256, 0, stream>>>(buf1, wp[2], bi[2], buf0);
  conv3x3_wmma<64, true ><<<nblk, 256, 0, stream>>>(buf0, wp[3], bi[3], d_out);
}
